// FastTransformerEncoderWrapper_13142599925857
// MI455X (gfx1250) — matrix operand order repaired and checked by reading, not compiled, not run on hardware
//
#include <hip/hip_runtime.h>
#include <hip/hip_bf16.h>
#include <stdint.h>
#include <math.h>

typedef __bf16 bf16;
typedef __attribute__((ext_vector_type(16))) __bf16 v16bf;
typedef __attribute__((ext_vector_type(8)))  __bf16 v8bf;
typedef __attribute__((ext_vector_type(8)))  float  v8f;

#define S_LEN  4096
#define DM     1024
#define NHD    16
#define DH     64
#define FF     4096
#define NLAYER 4
#define NBATCH 4
#define MTOT   (NBATCH * S_LEN)   // 16384 token rows

// ------------------------------------------------------------------
// CDNA5 async global->LDS (ASYNCcnt) helpers
// ------------------------------------------------------------------
__device__ __forceinline__ void async_b128(void* lds, const void* g) {
  // low 32 bits of a flat shared-aperture address are the LDS byte offset
  uint32_t off = (uint32_t)(uint64_t)(uintptr_t)lds;
  asm volatile("global_load_async_to_lds_b128 %0, %1, off" :: "v"(off), "v"(g) : "memory");
}
__device__ __forceinline__ void async_wait4() {   // next tile (4 ops) may stay in flight
  asm volatile("s_wait_asynccnt 4" ::: "memory");
}
__device__ __forceinline__ void async_wait0() {   // drain
  asm volatile("s_wait_asynccnt 0" ::: "memory");
}

// ------------------------------------------------------------------
// small elementwise / reduction kernels
// ------------------------------------------------------------------
__global__ __launch_bounds__(256) void cast_f32_bf16(const float* __restrict__ in,
                                                     bf16* __restrict__ out, long n) {
  long i = (long)blockIdx.x * blockDim.x + threadIdx.x;
  long stride = (long)gridDim.x * blockDim.x;
  for (; i < n; i += stride) out[i] = (bf16)in[i];
}

// actual_len[n] = S - sum(mask row); mask is bool (byte) per reference
__global__ __launch_bounds__(256) void len_kernel(const unsigned char* __restrict__ mask,
                                                  int* __restrict__ alen) {
  __shared__ int sred[256];
  int n = blockIdx.x, t = threadIdx.x, s = 0;
  for (int i = t; i < S_LEN; i += 256) s += (mask[n * S_LEN + i] != 0);
  sred[t] = s; __syncthreads();
  for (int off = 128; off > 0; off >>= 1) { if (t < off) sred[t] += sred[t + off]; __syncthreads(); }
  if (t == 0) alen[n] = S_LEN - sred[0];
}

// ------------------------------------------------------------------
// WMMA bf16 GEMM:  C[M,Nc] = A[M,K] * B[Nc,K]^T + bias, fused epilogue
// 256 threads = 8 waves (4x2); WG tile 128x128, wave tile 32x64
// (2 M-subtiles x 4 N-subtiles -> 8 v_wmma per 32-K step per wave).
// 3-stage LDS ring: async copy of tile k+1 overlaps WMMA on tile k;
// one s_barrier per step is race-free at depth 3 (writes of tile k+1
// only alias reads of tile k-2, which all waves finished before the
// barrier this wave already passed).
// ------------------------------------------------------------------
#define EPI_NONE      0   // f32 out
#define EPI_FEAT      1   // elu(x)+1 -> f32  (Q)
#define EPI_FEAT_MASK 2   // (elu(x)+1)*fmask -> f32 (K)
#define EPI_GELU_B16  3   // gelu(x) -> bf16  (FFN1)
#define EPI_RES       4   // x + res -> f32   (O-proj, FFN2)

__device__ __forceinline__ float feat_map(float x) { return x > 0.f ? x + 1.f : __expf(x); }
__device__ __forceinline__ float gelu_exact(float x) { return 0.5f * x * (1.f + erff(x * 0.70710678118f)); }

__device__ __forceinline__ void stage_tile(const bf16* __restrict__ A, const bf16* __restrict__ B,
                                           int K, int rowBase, int colBase, int k0,
                                           bf16 (*As)[32], bf16 (*Bs)[32], int t) {
  // A tile 128x32 (512 b128 chunks) + B tile 128x32 (512 chunks), 4 per thread
  int c = t, r = c >> 2, co = (c & 3) * 8;
  async_b128(&As[r][co], &A[(long)(rowBase + r) * K + k0 + co]);
  c = t + 256; r = c >> 2; co = (c & 3) * 8;
  async_b128(&As[r][co], &A[(long)(rowBase + r) * K + k0 + co]);
  c = t; r = c >> 2; co = (c & 3) * 8;
  async_b128(&Bs[r][co], &B[(long)(colBase + r) * K + k0 + co]);
  c = t + 256; r = c >> 2; co = (c & 3) * 8;
  async_b128(&Bs[r][co], &B[(long)(colBase + r) * K + k0 + co]);
}

template <int EPI>
__global__ void __launch_bounds__(256)
gemm_bf16_wmma(const bf16* __restrict__ A, const bf16* __restrict__ B,
               const float* __restrict__ bias,
               float* __restrict__ Cf, bf16* __restrict__ Cb,
               const float* __restrict__ res,
               int Nc, int K, const int* __restrict__ alen) {
  __shared__ __align__(16) bf16 As[3][128][32];   // 3-deep ring, 48 KB total
  __shared__ __align__(16) bf16 Bs[3][128][32];

  const int t    = threadIdx.x;
  const int lane = t & 31;
  const int wave = t >> 5;
  const int wr   = wave >> 1;            // 0..3 : 32-row band
  const int wc   = wave & 1;             // 0..1 : 64-col band
  const int rowBase = blockIdx.y * 128;
  const int colBase = blockIdx.x * 128;
  const int l15  = lane & 15;
  const int half = lane >> 4;            // 0 or 1

  v8f acc[2][4] = {{{}, {}, {}, {}}, {{}, {}, {}, {}}};

  const int nT = K >> 5;                 // K/32 steps
  stage_tile(A, B, K, rowBase, colBase, 0, As[0], Bs[0], t);

  for (int kt = 0; kt < nT; kt++) {
    const int cur = kt % 3;
    if (kt + 1 < nT) {
      stage_tile(A, B, K, rowBase, colBase, (kt + 1) << 5,
                 As[(kt + 1) % 3], Bs[(kt + 1) % 3], t);
      if (kt + 2 < nT) {  // pull the k+2 slab toward GL2 (global_prefetch_b8)
        __builtin_prefetch(&A[(long)(rowBase + (t >> 2)) * K + ((kt + 2) << 5) + (t & 3) * 8], 0, 1);
        __builtin_prefetch(&B[(long)(colBase + (t >> 2)) * K + ((kt + 2) << 5) + (t & 3) * 8], 0, 1);
      }
      async_wait4();   // tile kt complete; tile kt+1 still in flight
    } else {
      async_wait0();
    }
    __syncthreads();

    // ---- fragments (ISA 7.12.2 bf16 lane layouts) ----
    v16bf afrag[2], bfrag[4];
#pragma unroll
    for (int ms = 0; ms < 2; ms++) {
      int r = wr * 32 + ms * 16 + l15;
      v8bf lo = *(const v8bf*)&As[cur][r][8 * half];       // K = 8h..8h+7
      v8bf hi = *(const v8bf*)&As[cur][r][16 + 8 * half];  // K = 16+8h..
      afrag[ms] = __builtin_shufflevector(lo, hi, 0,1,2,3,4,5,6,7,8,9,10,11,12,13,14,15);
    }
#pragma unroll
    for (int ns = 0; ns < 4; ns++) {
      int r = wc * 64 + ns * 16 + l15;
      v8bf lo = *(const v8bf*)&Bs[cur][r][8 * half];       // K = 8h..8h+7
      v8bf hi = *(const v8bf*)&Bs[cur][r][16 + 8 * half];  // K = 16+8h..
      bfrag[ns] = __builtin_shufflevector(lo, hi, 0,1,2,3,4,5,6,7,8,9,10,11,12,13,14,15);
    }
#pragma unroll
    for (int ms = 0; ms < 2; ms++)
#pragma unroll
      for (int ns = 0; ns < 4; ns++)
        acc[ms][ns] = __builtin_amdgcn_wmma_f32_16x16x32_bf16(
            false, afrag[ms], false, bfrag[ns], (short)0, acc[ms][ns], false, false);
  }

  // ---- epilogue: D layout = lane(0-15): N=lane, vgpr e -> M=e; lane(16-31): M=8+e
#pragma unroll
  for (int ms = 0; ms < 2; ms++) {
#pragma unroll
    for (int ns = 0; ns < 4; ns++) {
      int col = colBase + wc * 64 + ns * 16 + l15;
      float bv = bias[col];
#pragma unroll
      for (int e = 0; e < 8; e++) {
        int row = rowBase + wr * 32 + ms * 16 + half * 8 + e;
        float v = acc[ms][ns][e] + bv;
        long idx = (long)row * Nc + col;
        if (EPI == EPI_NONE)      Cf[idx] = v;
        else if (EPI == EPI_FEAT) Cf[idx] = feat_map(v);
        else if (EPI == EPI_FEAT_MASK) {
          int n = row >> 12, s = row & (S_LEN - 1);
          Cf[idx] = (s < alen[n]) ? feat_map(v) : 0.f;
        }
        else if (EPI == EPI_GELU_B16) Cb[idx] = (bf16)gelu_exact(v);
        else if (EPI == EPI_RES)      Cf[idx] = v + res[idx];
      }
    }
  }
}

// ------------------------------------------------------------------
// KV[n,h,m,d] = sum_s V[n,s,h,m]*K[n,s,h,d];  ksum[n,h,d] = sum_s K
// one block per (n,h); ~2 GF/layer total -> VALU is fine
// ------------------------------------------------------------------
__global__ __launch_bounds__(256) void kv_ksum_kernel(const float* __restrict__ Kf,
                                                      const float* __restrict__ Vf,
                                                      float* __restrict__ KV,
                                                      float* __restrict__ ksum) {
  __shared__ __align__(16) float Ks[32][64];
  __shared__ __align__(16) float Vs[32][64];
  int nh = blockIdx.x, n = nh >> 4, h = nh & 15;
  int t = threadIdx.x;
  int m0 = t >> 2, dg = (t & 3) * 16;
  float acc[16];
#pragma unroll
  for (int i = 0; i < 16; i++) acc[i] = 0.f;
  float ksacc = 0.f;

  for (int c = 0; c < S_LEN / 32; c++) {
    int si = t >> 3, p = t & 7;
    long base = (((long)n * S_LEN + c * 32 + si) * NHD + h) * DH + p * 8;
    *(float4*)&Ks[si][p * 8]     = *(const float4*)&Kf[base];
    *(float4*)&Ks[si][p * 8 + 4] = *(const float4*)&Kf[base + 4];
    *(float4*)&Vs[si][p * 8]     = *(const float4*)&Vf[base];
    *(float4*)&Vs[si][p * 8 + 4] = *(const float4*)&Vf[base + 4];
    __syncthreads();
    if (t < 64)
      for (int s2 = 0; s2 < 32; s2++) ksacc += Ks[s2][t];
    for (int s2 = 0; s2 < 32; s2++) {
      float vm = Vs[s2][m0];
#pragma unroll
      for (int i = 0; i < 16; i++) acc[i] += vm * Ks[s2][dg + i];
    }
    __syncthreads();
  }
#pragma unroll
  for (int i = 0; i < 16; i++) KV[(long)nh * 4096 + (long)m0 * 64 + dg + i] = acc[i];
  if (t < 64) ksum[nh * 64 + t] = ksacc;
}

// ------------------------------------------------------------------
// attn[n,l,h,m] = (Q . KV[m,:]) / (Q . ksum + eps) -> bf16 for O-proj
// one wave per (token,head)
// ------------------------------------------------------------------
__global__ __launch_bounds__(256) void attn_kernel(const float* __restrict__ Qf,
                                                   const float* __restrict__ KV,
                                                   const float* __restrict__ ksum,
                                                   bf16* __restrict__ out) {
  long task = (long)blockIdx.x * 8 + (threadIdx.x >> 5);  // row*NH + h
  int lane = threadIdx.x & 31;
  long row = task >> 4;
  int h = (int)(task & 15);
  int n = (int)(row >> 12);
  int nh = n * 16 + h;

  const float* q = &Qf[row * DM + h * DH];
  float qr[64];
#pragma unroll
  for (int i = 0; i < 16; i++) {
    float4 v = ((const float4*)q)[i];
    qr[4 * i] = v.x; qr[4 * i + 1] = v.y; qr[4 * i + 2] = v.z; qr[4 * i + 3] = v.w;
  }
  const float* ks = &ksum[nh * 64];
  float part = qr[2 * lane] * ks[2 * lane] + qr[2 * lane + 1] * ks[2 * lane + 1];
#pragma unroll
  for (int m = 16; m >= 1; m >>= 1) part += __shfl_xor(part, m, 32);
  float z = 1.f / (part + 1e-6f);

  const float* kv = &KV[(long)nh * 4096];
#pragma unroll
  for (int mm = 0; mm < 2; mm++) {
    int m = 2 * lane + mm;
    const float* kvr = &kv[m * 64];
    float a = 0.f;
#pragma unroll
    for (int i = 0; i < 64; i++) a += qr[i] * kvr[i];
    out[row * DM + h * DH + m] = (bf16)(a * z);
  }
}

// ------------------------------------------------------------------
// LayerNorm over D=1024; optional residual input R; writes f32 + bf16
// ------------------------------------------------------------------
__global__ __launch_bounds__(256) void ln_kernel(const float* __restrict__ X,
                                                 const float* __restrict__ R,
                                                 const float* __restrict__ g,
                                                 const float* __restrict__ b,
                                                 float* __restrict__ outF,
                                                 bf16* __restrict__ outB) {
  __shared__ float red[8];
  int row = blockIdx.x, t = threadIdx.x;
  long base = (long)row * DM;
  float x[4];
#pragma unroll
  for (int i = 0; i < 4; i++) {
    int c = t + 256 * i;
    x[i] = X[base + c] + (R ? R[base + c] : 0.f);
  }
  float s = x[0] + x[1] + x[2] + x[3];
#pragma unroll
  for (int m = 16; m >= 1; m >>= 1) s += __shfl_xor(s, m, 32);
  if ((t & 31) == 0) red[t >> 5] = s;
  __syncthreads();
  float tot = 0.f;
  for (int i = 0; i < 8; i++) tot += red[i];
  float mu = tot * (1.f / DM);
  __syncthreads();
  float vs = 0.f;
#pragma unroll
  for (int i = 0; i < 4; i++) { float d = x[i] - mu; vs += d * d; }
#pragma unroll
  for (int m = 16; m >= 1; m >>= 1) vs += __shfl_xor(vs, m, 32);
  if ((t & 31) == 0) red[t >> 5] = vs;
  __syncthreads();
  float vt = 0.f;
  for (int i = 0; i < 8; i++) vt += red[i];
  float inv = rsqrtf(vt * (1.f / DM) + 1e-5f);
#pragma unroll
  for (int i = 0; i < 4; i++) {
    int c = t + 256 * i;
    float o = (x[i] - mu) * inv * g[c] + b[c];
    outF[base + c] = o;
    if (outB) outB[base + c] = (bf16)o;
  }
}

// ------------------------------------------------------------------
// host orchestration
// ------------------------------------------------------------------
extern "C" void kernel_launch(void* const* d_in, const int* in_sizes, int n_in,
                              void* d_out, int out_size, void* d_ws, size_t ws_size,
                              hipStream_t stream) {
  const float* src          = (const float*)d_in[0];
  const unsigned char* mask = (const unsigned char*)d_in[1];
  const float* Wq = (const float*)d_in[2];  const float* bq = (const float*)d_in[3];
  const float* Wk = (const float*)d_in[4];  const float* bk = (const float*)d_in[5];
  const float* Wv = (const float*)d_in[6];  const float* bv = (const float*)d_in[7];
  const float* Wo = (const float*)d_in[8];  const float* bo = (const float*)d_in[9];
  const float* W1 = (const float*)d_in[10]; const float* b1 = (const float*)d_in[11];
  const float* W2 = (const float*)d_in[12]; const float* b2 = (const float*)d_in[13];
  const float* g1 = (const float*)d_in[14]; const float* be1 = (const float*)d_in[15];
  const float* g2 = (const float*)d_in[16]; const float* be2 = (const float*)d_in[17];

  char* ws = (char*)d_ws;
  size_t off = 0;
  auto take = [&](size_t bytes) { char* p = ws + off; off += (bytes + 255) & ~(size_t)255; return p; };

  const long DD = (long)DM * DM, FD = (long)FF * DM;
  bf16* wqb = (bf16*)take(NLAYER * DD * 2);
  bf16* wkb = (bf16*)take(NLAYER * DD * 2);
  bf16* wvb = (bf16*)take(NLAYER * DD * 2);
  bf16* wob = (bf16*)take(NLAYER * DD * 2);
  bf16* w1b = (bf16*)take(NLAYER * FD * 2);
  bf16* w2b = (bf16*)take(NLAYER * FD * 2);
  const long ACT = (long)MTOT * DM;            // 16M elems
  float* Abuf = (float*)take(ACT * 4);          // y2 / next-x
  bf16*  Bbuf = (bf16*)take(ACT * 2);           // xb / attn_b
  float* Cbuf = (float*)take(ACT * 4);          // Q then y
  float* Dbuf = (float*)take(ACT * 4);          // K then x2
  float* Ebuf = (float*)take(ACT * 4);          // V; also reused as x2 bf16
  bf16*  Fbuf = (bf16*)take((long)MTOT * FF * 2); // FFN hidden (bf16)
  float* KVb  = (float*)take(64 * 4096 * 4);
  float* ksb  = (float*)take(64 * 64 * 4);
  int*   alen = (int*)take(256);
  bf16*  x2b  = (bf16*)Ebuf;                    // alias: V dead when x2b is live

  // bf16 weight casts + initial activation cast + mask lengths
  cast_f32_bf16<<<4096, 256, 0, stream>>>(Wq, wqb, NLAYER * DD);
  cast_f32_bf16<<<4096, 256, 0, stream>>>(Wk, wkb, NLAYER * DD);
  cast_f32_bf16<<<4096, 256, 0, stream>>>(Wv, wvb, NLAYER * DD);
  cast_f32_bf16<<<4096, 256, 0, stream>>>(Wo, wob, NLAYER * DD);
  cast_f32_bf16<<<4096, 256, 0, stream>>>(W1, w1b, NLAYER * FD);
  cast_f32_bf16<<<4096, 256, 0, stream>>>(W2, w2b, NLAYER * FD);
  cast_f32_bf16<<<4096, 256, 0, stream>>>(src, Bbuf, ACT);
  len_kernel<<<NBATCH, 256, 0, stream>>>(mask, alen);

  const float* x = src;  // f32 residual source for layer 0
  dim3 blk(256);
  dim3 gD(DM / 128, MTOT / 128);   // Nc=1024 -> (8,128)
  dim3 gF(FF / 128, MTOT / 128);   // Nc=4096 -> (32,128)

  for (int l = 0; l < NLAYER; l++) {
    const bf16 *wq = wqb + l * DD, *wk = wkb + l * DD, *wv = wvb + l * DD, *wo = wob + l * DD;
    const bf16 *w1 = w1b + l * FD, *w2 = w2b + l * FD;
    const float *bql = bq + l * DM, *bkl = bk + l * DM, *bvl = bv + l * DM, *bol = bo + l * DM;
    const float *b1l = b1 + l * FF, *b2l = b2 + l * DM;
    const float *g1l = g1 + l * DM, *be1l = be1 + l * DM, *g2l = g2 + l * DM, *be2l = be2 + l * DM;

    // Q/K/V projections with fused feature map / mask
    gemm_bf16_wmma<EPI_FEAT>     <<<gD, blk, 0, stream>>>(Bbuf, wq, bql, Cbuf, nullptr, nullptr, DM, DM, nullptr);
    gemm_bf16_wmma<EPI_FEAT_MASK><<<gD, blk, 0, stream>>>(Bbuf, wk, bkl, Dbuf, nullptr, nullptr, DM, DM, alen);
    gemm_bf16_wmma<EPI_NONE>     <<<gD, blk, 0, stream>>>(Bbuf, wv, bvl, Ebuf, nullptr, nullptr, DM, DM, nullptr);

    // linear attention: KV/ksum then normalized output (bf16) into Bbuf
    kv_ksum_kernel<<<64, blk, 0, stream>>>(Dbuf, Ebuf, KVb, ksb);
    attn_kernel<<<MTOT * NHD / 8, blk, 0, stream>>>(Cbuf, KVb, ksb, Bbuf);

    // O projection + residual -> y (Cbuf); LN1 -> x2 (Dbuf f32, x2b bf16)
    gemm_bf16_wmma<EPI_RES><<<gD, blk, 0, stream>>>(Bbuf, wo, bol, Cbuf, nullptr, x, DM, DM, nullptr);
    ln_kernel<<<MTOT, blk, 0, stream>>>(Cbuf, nullptr, g1l, be1l, Dbuf, x2b);

    // FFN: gelu(x2 W1^T + b1) W2^T + b2 + x2 -> y2 (Abuf); LN2 -> next x
    gemm_bf16_wmma<EPI_GELU_B16><<<gF, blk, 0, stream>>>(x2b, w1, b1l, nullptr, Fbuf, nullptr, FF, DM, nullptr);
    gemm_bf16_wmma<EPI_RES>     <<<gD, blk, 0, stream>>>(Fbuf, w2, b2l, Abuf, nullptr, Dbuf, DM, FF, nullptr);

    float* xnext = (l == NLAYER - 1) ? (float*)d_out : Abuf;
    ln_kernel<<<MTOT, blk, 0, stream>>>(Abuf, Dbuf, g2l, be2l, xnext, Bbuf);
    x = Abuf;
  }
}
